// Transformer_257698037867
// MI455X (gfx1250) — matrix operand order repaired and checked by reading, not compiled, not run on hardware
//
#include <hip/hip_runtime.h>
#include <hip/hip_bf16.h>

// ---------------------------------------------------------------------------
// Transformer forward for gfx1250 (MI455X): bf16 WMMA everywhere.
//   DIM=1024, DEPTH=6, HEADS=16 (hd=64), MLP=4096, B=4, N=2048  -> M=8192 rows
// Weights are converted fp32->bf16 AND transposed once per launch, so every
// GEMM inner loop is pure {ds_load_b128 -> v_wmma_f32_16x16x32_bf16}.
// ---------------------------------------------------------------------------

#define DIM   1024
#define DEPTH 6
#define HEADS 16
#define HD    64
#define MLP   4096
#define BSZ   4
#define SEQ   2048
#define MROWS (BSZ * SEQ)          // 8192

typedef __attribute__((ext_vector_type(16))) __bf16 v16bf;
typedef __attribute__((ext_vector_type(8)))  float  v8f;

struct FragBF {
    union { int i[8]; v16bf v; };
};

__device__ __forceinline__ unsigned short f32_to_bf16(float f) {
    union { float f; unsigned int u; } c; c.f = f;
    unsigned int r = c.u + 0x7FFFu + ((c.u >> 16) & 1u);   // round-nearest-even
    return (unsigned short)(r >> 16);
}

// A-operand fragment (16x32 bf16): lanes 0-15 hold K=0..15,
// lanes 16-31 hold K=16..31 (matches B); caller bakes row + half-offset into p.
__device__ __forceinline__ void load_fragA(FragBF& f, const unsigned short* p) {
    const int* q0 = (const int*)(p);
    const int* q1 = (const int*)(p + 8);
#pragma unroll
    for (int j = 0; j < 4; ++j) { f.i[j] = q0[j]; f.i[4 + j] = q1[j]; }
}

// B-operand fragment (32x16 bf16): per lane 16 contiguous bf16 (K half-block).
__device__ __forceinline__ void load_fragB(FragBF& f, const unsigned short* p) {
    const int* q = (const int*)(p);
#pragma unroll
    for (int j = 0; j < 8; ++j) f.i[j] = q[j];
}

__device__ __forceinline__ v8f wmma_bf16(const FragBF& a, const FragBF& b, v8f c) {
    return __builtin_amdgcn_wmma_f32_16x16x32_bf16(false, a.v, false, b.v,
                                                   (short)0, c, false, false);
}

__device__ __forceinline__ float gelu_exact(float x) {
    return 0.5f * x * (1.0f + erff(x * 0.70710678118654752f));
}

// ---------------------------------------------------------------------------
// Weight prep (once per launch): fp32 [K][Nc] -> bf16 transposed [Nc][K].
// LDS-tiled 32x32 so both global read and write are coalesced.
// ---------------------------------------------------------------------------
__global__ __launch_bounds__(256)
void transpose_bf16_kernel(const float* __restrict__ in,
                           unsigned short* __restrict__ out, int K, int Nc) {
    __shared__ unsigned short t[32][33];       // +1 pad: no bank conflicts
    const int d = blockIdx.z;
    in  += (size_t)d * K * Nc;
    out += (size_t)d * Nc * K;
    const int n0 = blockIdx.x * 32;
    const int k0 = blockIdx.y * 32;
    const int tx = threadIdx.x;                // 0..31
    const int ty = threadIdx.y;                // 0..7
#pragma unroll
    for (int j = 0; j < 4; ++j) {
        const int k = ty + j * 8;
        t[k][tx] = f32_to_bf16(in[(size_t)(k0 + k) * Nc + n0 + tx]);
    }
    __syncthreads();
#pragma unroll
    for (int j = 0; j < 4; ++j) {
        const int n = ty + j * 8;
        out[(size_t)(n0 + n) * K + k0 + tx] = t[tx][n];
    }
}

// ---------------------------------------------------------------------------
// LayerNorm over DIM=1024; one block per row; bf16 or f32 output.
// ---------------------------------------------------------------------------
template <bool OUT_BF16>
__global__ __launch_bounds__(256)
void ln_kernel(const float* __restrict__ x, const float* __restrict__ g,
               const float* __restrict__ bb, void* __restrict__ out) {
    const int row = blockIdx.x;
    const int tid = threadIdx.x;
    const float* xr = x + (size_t)row * DIM;
    float v[4];
    float s = 0.f, sq = 0.f;
#pragma unroll
    for (int j = 0; j < 4; ++j) {
        v[j] = xr[tid + 256 * j];
        s += v[j]; sq += v[j] * v[j];
    }
#pragma unroll
    for (int m = 1; m < 32; m <<= 1) {
        s  += __shfl_xor(s,  m, 32);
        sq += __shfl_xor(sq, m, 32);
    }
    __shared__ float sh[16];
    const int wid = tid >> 5, lane = tid & 31;
    if (lane == 0) { sh[wid] = s; sh[8 + wid] = sq; }
    __syncthreads();
    s = 0.f; sq = 0.f;
#pragma unroll
    for (int w = 0; w < 8; ++w) { s += sh[w]; sq += sh[8 + w]; }
    const float mean = s * (1.f / DIM);
    const float var  = sq * (1.f / DIM) - mean * mean;
    const float rstd = rsqrtf(var + 1e-5f);
#pragma unroll
    for (int j = 0; j < 4; ++j) {
        const int c = tid + 256 * j;
        const float y = (v[j] - mean) * rstd * g[c] + bb[c];
        if (OUT_BF16) ((unsigned short*)out)[(size_t)row * DIM + c] = f32_to_bf16(y);
        else          ((float*)out)[(size_t)row * DIM + c] = y;
    }
}

// ---------------------------------------------------------------------------
// x += attn_out ; x_bf16 = bf16(x)
// ---------------------------------------------------------------------------
__global__ void resid_kernel(float* __restrict__ x, const float* __restrict__ a,
                             unsigned short* __restrict__ xb, size_t n) {
    size_t stride = (size_t)gridDim.x * blockDim.x;
    for (size_t i = (size_t)blockIdx.x * blockDim.x + threadIdx.x; i < n; i += stride) {
        float v = x[i] + a[i];
        x[i] = v;
        xb[i] = f32_to_bf16(v);
    }
}

// ---------------------------------------------------------------------------
// bf16 GEMM: out[M,Nc] = epilogue(A[M,K] @ W[K,Nc]), W given TRANSPOSED
// as Wt[Nc][K] bf16 so LDS staging is contiguous (no in-loop transpose).
//   MODE 0: store bf16, no bias                       (QKV)
//   MODE 1: +bias, GELU, store bf16                   (FFN1)
//   MODE 2: +bias, +resid (f32), store f32            (FFN2)
// Block: 256 threads = 8 waves; tile 256x128x32; wave tile 64x64:
// 16 WMMAs per 16 ds_load_b128 per k-step.
// ---------------------------------------------------------------------------
#define BM 256
#define BN 128
#define BK 32

template <int MODE>
__global__ __launch_bounds__(256)
void gemm_bf16_kernel(const unsigned short* __restrict__ A,
                      const unsigned short* __restrict__ Wt,
                      const float* __restrict__ bias,
                      const float* __restrict__ resid,
                      void* __restrict__ out, int K, int Nc) {
    __shared__ unsigned short As[BM * BK];   // [256][32] row-major (m, k)
    __shared__ unsigned short Bt[BN * BK];   // [128][32] row-major (n, k)

    const int tid  = threadIdx.x;
    const int lane = tid & 31;
    const int wid  = tid >> 5;
    const int waveM = wid >> 1;              // 0..3 -> 64-row strips
    const int waveN = wid & 1;               // 0..1 -> 64-col strips
    const int ln = lane & 15;
    const int hl = lane >> 4;

    const size_t blockRow = (size_t)blockIdx.y * BM;
    const size_t blockCol = (size_t)blockIdx.x * BN;

    const v8f z8 = {0.f, 0.f, 0.f, 0.f, 0.f, 0.f, 0.f, 0.f};
    v8f acc[4][4];
#pragma unroll
    for (int tm = 0; tm < 4; ++tm)
#pragma unroll
        for (int tn = 0; tn < 4; ++tn) acc[tm][tn] = z8;

    // staging: A -> thread t copies one 64B row slice; Wt -> 32B half-rows
    const unsigned short* Aptr = A + (blockRow + tid) * (size_t)K;
    const unsigned short* Wptr = Wt + (blockCol + (tid >> 1)) * (size_t)K + (tid & 1) * 16;

    for (int k0 = 0; k0 < K; k0 += BK) {
        __syncthreads();
        {
            const int4* ag = (const int4*)(Aptr + k0);
            int4* ad = (int4*)(As + tid * BK);
            ad[0] = ag[0]; ad[1] = ag[1]; ad[2] = ag[2]; ad[3] = ag[3];

            const int4* wg = (const int4*)(Wptr + k0);
            int4* wd = (int4*)(Bt + (tid >> 1) * BK + (tid & 1) * 16);
            wd[0] = wg[0]; wd[1] = wg[1];
        }
        if (k0 + BK < K) {
            __builtin_prefetch(Aptr + k0 + BK, 0, 0);
            __builtin_prefetch(Wptr + k0 + BK, 0, 0);
        }
        __syncthreads();

        FragBF af[4], bfr[4];
#pragma unroll
        for (int tm = 0; tm < 4; ++tm)
            load_fragA(af[tm], As + (waveM * 64 + tm * 16 + ln) * BK + hl * 16);
#pragma unroll
        for (int tn = 0; tn < 4; ++tn)
            load_fragB(bfr[tn], Bt + (waveN * 64 + tn * 16 + ln) * BK + hl * 16);
#pragma unroll
        for (int tm = 0; tm < 4; ++tm)
#pragma unroll
            for (int tn = 0; tn < 4; ++tn)
                acc[tm][tn] = wmma_bf16(af[tm], bfr[tn], acc[tm][tn]);
    }

    // ---- epilogue (C/D layout: row = 8*(lane>=16)+r, col = lane%16)
    const size_t rowBase = blockRow + waveM * 64;
    const size_t colBase = blockCol + waveN * 64;
#pragma unroll
    for (int tn = 0; tn < 4; ++tn) {
        const size_t col = colBase + tn * 16 + ln;
        float bval = 0.f;
        if (MODE >= 1) bval = bias[col];
#pragma unroll
        for (int tm = 0; tm < 4; ++tm) {
#pragma unroll
            for (int r = 0; r < 8; ++r) {
                const size_t row = rowBase + tm * 16 + hl * 8 + r;
                float v = acc[tm][tn][r];
                if (MODE == 1) v = gelu_exact(v + bval);
                if (MODE == 2) v = v + bval + resid[row * (size_t)Nc + col];
                if (MODE == 2) ((float*)out)[row * (size_t)Nc + col] = v;
                else ((unsigned short*)out)[row * (size_t)Nc + col] = f32_to_bf16(v);
            }
        }
    }
}

// ---------------------------------------------------------------------------
// Flash attention: one block = one (batch, head, 128-query tile).
// 8 waves; each wave owns 16 query rows. KV tiles of 64 staged in LDS.
// S = Q K^T (WMMA), online softmax, O += P V (WMMA). scale = DIM^-0.5.
// ---------------------------------------------------------------------------
__global__ __launch_bounds__(256)
void attn_kernel(const unsigned short* __restrict__ qkv, float* __restrict__ out) {
    __shared__ unsigned short Kt[64 * 64];      // [kv][hd]
    __shared__ unsigned short Vt[64 * 64];      // [hd][kv] (transposed)
    __shared__ unsigned short Ps[8][16 * 64];   // per-wave P tile [16][64]

    const int tid  = threadIdx.x;
    const int lane = tid & 31;
    const int wid  = tid >> 5;
    const int ln = lane & 15;
    const int hl = lane >> 4;

    const int b = blockIdx.z;
    const int h = blockIdx.y;
    const int qbase = blockIdx.x * 128;
    const float scale = 0.03125f;               // 1/sqrt(1024)

    // ---- load Q fragments once (A operand, 16 rows x 64 hd -> 2 k-steps)
    FragBF qf[2];
    {
        const int qrow = qbase + wid * 16 + ln;
        const unsigned short* qptr =
            qkv + ((size_t)(b * SEQ + qrow)) * (3 * DIM) + h * HD;
#pragma unroll
        for (int s = 0; s < 2; ++s)
            load_fragA(qf[s], qptr + s * 32 + hl * 16);
    }

    const v8f z8 = {0.f, 0.f, 0.f, 0.f, 0.f, 0.f, 0.f, 0.f};
    v8f o[4];
#pragma unroll
    for (int t = 0; t < 4; ++t) o[t] = z8;
    float mprev[8], lsum[8];
#pragma unroll
    for (int r = 0; r < 8; ++r) { mprev[r] = -3.0e38f; lsum[r] = 0.f; }

    const int sr = tid >> 2;                 // 0..63 staging row
    const int sc = (tid & 3) * 16;           // 0..48 staging col

    for (int kv0 = 0; kv0 < SEQ; kv0 += 64) {
        __syncthreads();
        // ---- stage K row-major, V transposed
        {
            const unsigned short* kg =
                qkv + ((size_t)(b * SEQ + kv0 + sr)) * (3 * DIM) + DIM + h * HD + sc;
            int4* kd = (int4*)(Kt + sr * 64 + sc);
            kd[0] = ((const int4*)kg)[0];
            kd[1] = ((const int4*)kg)[1];

            const unsigned short* vg =
                qkv + ((size_t)(b * SEQ + kv0 + sr)) * (3 * DIM) + 2 * DIM + h * HD + sc;
            alignas(16) unsigned short tv[16];
            *(int4*)(tv)     = ((const int4*)vg)[0];
            *(int4*)(tv + 8) = ((const int4*)vg)[1];
#pragma unroll
            for (int j = 0; j < 16; ++j)
                Vt[(sc + j) * 64 + sr] = tv[j];
        }
        __syncthreads();

        // ---- S = Q K^T  (4 kv tiles of 16, 2 k-steps over hd)
        v8f st[4];
#pragma unroll
        for (int tn = 0; tn < 4; ++tn) st[tn] = z8;
        FragBF kf;
#pragma unroll
        for (int tn = 0; tn < 4; ++tn) {
#pragma unroll
            for (int s = 0; s < 2; ++s) {
                load_fragB(kf, Kt + (tn * 16 + ln) * 64 + hl * 16 + s * 32);
                st[tn] = wmma_bf16(qf[s], kf, st[tn]);
            }
        }
#pragma unroll
        for (int tn = 0; tn < 4; ++tn) st[tn] *= scale;

        // ---- online softmax (rows live in half-waves; shfl within 16 lanes)
        float corr[8];
#pragma unroll
        for (int r = 0; r < 8; ++r) {
            float mx = st[0][r];
            mx = fmaxf(mx, st[1][r]);
            mx = fmaxf(mx, st[2][r]);
            mx = fmaxf(mx, st[3][r]);
#pragma unroll
            for (int m = 1; m < 16; m <<= 1)
                mx = fmaxf(mx, __shfl_xor(mx, m, 32));
            const float mnew = fmaxf(mprev[r], mx);
            corr[r] = __expf(mprev[r] - mnew);
            float rsum = 0.f;
#pragma unroll
            for (int tn = 0; tn < 4; ++tn) {
                const float p = __expf(st[tn][r] - mnew);
                st[tn][r] = p;
                rsum += p;
            }
#pragma unroll
            for (int m = 1; m < 16; m <<= 1)
                rsum += __shfl_xor(rsum, m, 32);
            lsum[r] = lsum[r] * corr[r] + rsum;
            mprev[r] = mnew;
        }
#pragma unroll
        for (int t = 0; t < 4; ++t)
#pragma unroll
            for (int r = 0; r < 8; ++r) o[t][r] *= corr[r];

        // ---- P (fp32 C-layout) -> bf16 A-layout via per-wave LDS scratch
        unsigned short* pw = &Ps[wid][0];
#pragma unroll
        for (int tn = 0; tn < 4; ++tn)
#pragma unroll
            for (int r = 0; r < 8; ++r)
                pw[(hl * 8 + r) * 64 + tn * 16 + ln] = f32_to_bf16(st[tn][r]);
        // same-wave LDS RAW: DS ops are in-order; compiler sees the alias.

        FragBF pf[2], vf;
#pragma unroll
        for (int s = 0; s < 2; ++s)
            load_fragA(pf[s], pw + ln * 64 + s * 32 + hl * 16);
#pragma unroll
        for (int t = 0; t < 4; ++t) {
#pragma unroll
            for (int s = 0; s < 2; ++s) {
                load_fragB(vf, Vt + (t * 16 + ln) * 64 + hl * 16 + s * 32);
                o[t] = wmma_bf16(pf[s], vf, o[t]);
            }
        }
    }

    // ---- normalize and store O (fp32)
#pragma unroll
    for (int r = 0; r < 8; ++r) {
        const float inv = 1.0f / lsum[r];
        const size_t row = (size_t)(b * SEQ + qbase + wid * 16 + hl * 8 + r);
#pragma unroll
        for (int t = 0; t < 4; ++t)
            out[row * DIM + h * HD + t * 16 + ln] = o[t][r] * inv;
    }
}

// ---------------------------------------------------------------------------
// Launcher
// ---------------------------------------------------------------------------
extern "C" void kernel_launch(void* const* d_in, const int* in_sizes, int n_in,
                              void* d_out, int out_size, void* d_ws, size_t ws_size,
                              hipStream_t stream) {
    (void)in_sizes; (void)n_in; (void)out_size; (void)ws_size;

    const float* x_in = (const float*)d_in[0];
    const float* ln_g = (const float*)d_in[1];
    const float* ln_b = (const float*)d_in[2];
    const float* Wqkv = (const float*)d_in[3];
    const float* W1   = (const float*)d_in[4];
    const float* b1   = (const float*)d_in[5];
    const float* W2   = (const float*)d_in[6];
    const float* b2   = (const float*)d_in[7];
    const float* fg   = (const float*)d_in[8];
    const float* fb   = (const float*)d_in[9];

    // workspace layout
    char* ws = (char*)d_ws;
    const size_t N_QKV = (size_t)DEPTH * DIM * 3 * DIM;
    const size_t N_W1  = (size_t)DEPTH * DIM * MLP;
    const size_t N_W2  = (size_t)DEPTH * MLP * DIM;
    const size_t N_X   = (size_t)MROWS * DIM;
    const size_t N_QKVACT = (size_t)MROWS * 3 * DIM;
    const size_t N_H1  = (size_t)MROWS * MLP;

    unsigned short* wqkv_bf = (unsigned short*)ws;               ws += N_QKV * 2;
    unsigned short* w1_bf   = (unsigned short*)ws;               ws += N_W1 * 2;
    unsigned short* w2_bf   = (unsigned short*)ws;               ws += N_W2 * 2;
    float*          x_f32   = (float*)ws;                        ws += N_X * 4;
    unsigned short* x_bf    = (unsigned short*)ws;               ws += N_X * 2;
    unsigned short* h_bf    = (unsigned short*)ws;               ws += N_X * 2;
    unsigned short* qkv_bf  = (unsigned short*)ws;               ws += N_QKVACT * 2;
    float*          attn_f  = (float*)ws;                        ws += N_X * 4;
    unsigned short* h1_bf   = (unsigned short*)ws;               ws += N_H1 * 2;

    // one-time weight prep: fp32 [K][Nc] -> bf16 transposed [Nc][K]
    transpose_bf16_kernel<<<dim3(3 * DIM / 32, DIM / 32, DEPTH), dim3(32, 8), 0, stream>>>(
        Wqkv, wqkv_bf, DIM, 3 * DIM);
    transpose_bf16_kernel<<<dim3(MLP / 32, DIM / 32, DEPTH), dim3(32, 8), 0, stream>>>(
        W1, w1_bf, DIM, MLP);
    transpose_bf16_kernel<<<dim3(DIM / 32, MLP / 32, DEPTH), dim3(32, 8), 0, stream>>>(
        W2, w2_bf, MLP, DIM);

    hipMemcpyAsync(x_f32, x_in, N_X * sizeof(float), hipMemcpyDeviceToDevice, stream);

    for (int i = 0; i < DEPTH; ++i) {
        // h = LN(x)
        ln_kernel<true><<<MROWS, 256, 0, stream>>>(
            x_f32, ln_g + (size_t)i * DIM, ln_b + (size_t)i * DIM, h_bf);

        // qkv = h @ Wqkv
        gemm_bf16_kernel<0><<<dim3(3 * DIM / BN, MROWS / BM), 256, 0, stream>>>(
            h_bf, wqkv_bf + (size_t)i * DIM * 3 * DIM, nullptr, nullptr,
            qkv_bf, DIM, 3 * DIM);

        // attention
        attn_kernel<<<dim3(SEQ / 128, HEADS, BSZ), 256, 0, stream>>>(qkv_bf, attn_f);

        // x += attn; x_bf = bf16(x)
        resid_kernel<<<4096, 256, 0, stream>>>(x_f32, attn_f, x_bf, N_X);

        // h1 = gelu(x @ W1 + b1)
        gemm_bf16_kernel<1><<<dim3(MLP / BN, MROWS / BM), 256, 0, stream>>>(
            x_bf, w1_bf + (size_t)i * DIM * MLP, b1 + (size_t)i * MLP, nullptr,
            h1_bf, DIM, MLP);

        // x = h1 @ W2 + b2 + x
        gemm_bf16_kernel<2><<<dim3(DIM / BN, MROWS / BM), 256, 0, stream>>>(
            h1_bf, w2_bf + (size_t)i * MLP * DIM, b2 + (size_t)i * DIM, x_f32,
            x_f32, MLP, DIM);
    }

    // final LN -> fp32 output
    ln_kernel<false><<<MROWS, 256, 0, stream>>>(x_f32, fg, fb, d_out);
}
